// vmamba_tiny_17085379904464
// MI455X (gfx1250) — compile-verified, matrix operand order checked by reading, not run on hardware
//
#include <hip/hip_runtime.h>
#include <hip/hip_bf16.h>
#include <utility>

// ---------------------------------------------------------------------------
// VMamba-tiny forward for MI455X (gfx1250, wave32, WMMA).
// All channel-mixing einsums -> bf16 WMMA GEMM (f32 accumulate) with fused
// bias / GELU / softplus / residual epilogues. Selective scan specialized to
// N_STATE=1 scalar recurrence; reverse directions run the scan backwards
// instead of materializing flipped tensors.
// ---------------------------------------------------------------------------

typedef __attribute__((ext_vector_type(16))) __bf16 v16bf;
typedef __attribute__((ext_vector_type(8)))  float  v8f;

#define TB 256
#define BM 128
#define BN 64
#define BK 32
#define LDT 48   // padded LDS row (bf16 elems): 96B rows keep uint4 loads aligned

__device__ __forceinline__ unsigned bfbits(float f) {
  union { float f; unsigned u; } a; a.f = f;
  return (a.u + 0x7fffu + ((a.u >> 16) & 1u)) >> 16;         // RNE
}
__device__ __forceinline__ unsigned packbf(float lo, float hi) {
  return bfbits(lo) | (bfbits(hi) << 16);
}
__device__ __forceinline__ __bf16 f2bf(float f) {
  unsigned r = bfbits(f);
  union { unsigned short s; __bf16 b; } c; c.s = (unsigned short)r;
  return c.b;
}
__device__ __forceinline__ float geluf(float x) {
  float x3 = x * x * x;
  return 0.5f * x * (1.0f + tanhf(0.79788456080286535588f * (x + 0.044715f * x3)));
}
__device__ __forceinline__ float softplusf(float x) {
  return (x > 20.f) ? x : log1pf(expf(x));
}
__device__ __forceinline__ float siluf(float x) { return x / (1.f + expf(-x)); }

// ---------------------------------------------------------------------------
// WMMA GEMM: Y(m,n) = act( sum_k X(m,k)*W(k,n) + bias(n) ) + res(m,n)
// X is channel-major f32: X[(m/HW)*xbs + k*HW + (m%HW)]; Y likewise with ybs.
// W is bf16 row-major K x N (pre-converted). act: 0=none 1=gelu 2=softplus.
//
// Main K-loop is completely mask-free: per-thread (b,hw) staging coordinates
// are loop-invariant and clamped once; garbage in m>=M rows / n>=N cols is
// benign (those D rows/cols are never stored), and only the final partial
// K-tile needs k-masking (handled in a separate tail tile). EXEC stays
// all-ones throughout, as WMMA requires.
// ---------------------------------------------------------------------------
__global__ __launch_bounds__(TB)
void k_gemm_wmma(const float* __restrict__ X, const __bf16* __restrict__ W,
                 const float* __restrict__ bias, const float* __restrict__ res,
                 float* __restrict__ Y,
                 int M, int N, int K, int HW, int xbs, int ybs, int act)
{
  __shared__ alignas(16) __bf16 sm[BM * LDT + BN * LDT];
  __bf16* As = sm;
  __bf16* Bs = sm + BM * LDT;

  const int tid  = threadIdx.x;
  const int m0   = blockIdx.x * BM;
  const int n0   = blockIdx.y * BN;
  const int lane = tid & 31, wave = tid >> 5;
  const int wm   = wave >> 1, wn = wave & 1;       // 4x2 wave grid -> 32x32/wave
  const int half = lane >> 4, lm = lane & 15;

  // ---- K-loop-invariant staging coordinates ----
  // A tile: thread stages column m = tid&127, contiguous rows k = 16*(tid>>7)+i
  const int am   = tid & (BM - 1);
  const int ak0  = (tid >> 7) * 16;                // 0 or 16
  const int agm  = m0 + am;
  const int agmc = (agm < M) ? agm : (M - 1);      // clamp: OOB rows are benign
  const int ab   = agmc / HW;
  const int ahw  = agmc - ab * HW;
  const float* __restrict__ Xp = X + (size_t)ab * xbs + ahw;

  // B tile: thread stages column n = tid&63, contiguous rows k = 8*(tid>>6)+i
  const int bn   = tid & (BN - 1);
  const int bk0  = (tid >> 6) * 8;                 // 0,8,16,24
  const int bgn  = n0 + bn;
  const int bgnc = (bgn < N) ? bgn : (N - 1);      // clamp: OOB cols are benign
  const unsigned short* __restrict__ Wp = (const unsigned short*)W + bgnc;

  v8f acc[2][2] = {};

  // 4 WMMAs on the staged tiles (shared by main loop and tail)
  auto compute = [&]() {
    union Frag { v16bf v; uint4 q[2]; };
    Frag a[2], b[2];
#pragma unroll
    for (int mi = 0; mi < 2; ++mi) {
      const __bf16* p = &As[(wm * 32 + mi * 16 + lm) * LDT + 8 * half];
      a[mi].q[0] = *(const uint4*)p;
      a[mi].q[1] = *(const uint4*)(p + 16);
    }
#pragma unroll
    for (int ni = 0; ni < 2; ++ni) {
      const __bf16* p = &Bs[(wn * 32 + ni * 16 + lm) * LDT + 8 * half];
      b[ni].q[0] = *(const uint4*)p;
      b[ni].q[1] = *(const uint4*)(p + 16);
    }
#pragma unroll
    for (int mi = 0; mi < 2; ++mi)
#pragma unroll
      for (int ni = 0; ni < 2; ++ni)
        acc[mi][ni] = __builtin_amdgcn_wmma_f32_16x16x32_bf16(
            false, a[mi].v, false, b[ni].v, (short)0, acc[mi][ni], false, false);
  };

  const int kfull = (K / BK) * BK;

  for (int kk = 0; kk < kfull; kk += BK) {
    // Mask-free staging: pack bf16 pairs, one ds_store_b32 per pair
#pragma unroll
    for (int i = 0; i < 16; i += 2) {
      int k = ak0 + i;
      float v0 = Xp[(kk + k) * HW];
      float v1 = Xp[(kk + k + 1) * HW];
      *(unsigned*)&As[am * LDT + k] = packbf(v0, v1);
    }
#pragma unroll
    for (int i = 0; i < 8; i += 2) {
      int k = bk0 + i;
      unsigned w0 = Wp[(kk + k) * N];
      unsigned w1 = Wp[(kk + k + 1) * N];
      *(unsigned*)&Bs[bn * LDT + k] = w0 | (w1 << 16);
    }
    if (kk + BK < K)  // lowers to global_prefetch_b8 on gfx1250
      __builtin_prefetch(Wp + (kk + BK) * N, 0, 1);
    __syncthreads();
    compute();
    __syncthreads();
  }

  if (kfull < K) {   // tail tile: zero-fill k >= K (only k pollutes the sum)
    int kk = kfull;
#pragma unroll
    for (int i = 0; i < 16; i += 2) {
      int k = ak0 + i;
      int g0 = kk + k, g1 = g0 + 1;
      float v0 = Xp[((g0 < K) ? g0 : (K - 1)) * HW];
      float v1 = Xp[((g1 < K) ? g1 : (K - 1)) * HW];
      v0 = (g0 < K) ? v0 : 0.f;
      v1 = (g1 < K) ? v1 : 0.f;
      *(unsigned*)&As[am * LDT + k] = packbf(v0, v1);
    }
#pragma unroll
    for (int i = 0; i < 8; i += 2) {
      int k = bk0 + i;
      int g0 = kk + k, g1 = g0 + 1;
      unsigned w0 = Wp[((g0 < K) ? g0 : (K - 1)) * N];
      unsigned w1 = Wp[((g1 < K) ? g1 : (K - 1)) * N];
      w0 = (g0 < K) ? w0 : 0u;
      w1 = (g1 < K) ? w1 : 0u;
      *(unsigned*)&Bs[bn * LDT + k] = w0 | (w1 << 16);
    }
    __syncthreads();
    compute();
    __syncthreads();
  }

  // Epilogue: C/D layout -> lane gives N=lm, VGPR r gives M = r + 8*half
  int   gncol[2];
  float bv[2];
#pragma unroll
  for (int ni = 0; ni < 2; ++ni) {
    gncol[ni] = n0 + wn * 32 + ni * 16 + lm;
    bv[ni] = (bias && gncol[ni] < N) ? bias[gncol[ni]] : 0.f;
  }
#pragma unroll
  for (int mi = 0; mi < 2; ++mi) {
#pragma unroll
    for (int r = 0; r < 8; ++r) {
      int gm = m0 + wm * 32 + mi * 16 + 8 * half + r;
      if (gm >= M) continue;
      int bb = gm / HW, hw = gm - bb * HW;
      size_t rowoff = (size_t)bb * ybs + hw;
#pragma unroll
      for (int ni = 0; ni < 2; ++ni) {
        if (gncol[ni] >= N) continue;
        float v = acc[mi][ni][r] + bv[ni];
        if (act == 1)      v = geluf(v);
        else if (act == 2) v = softplusf(v);
        size_t o = rowoff + (size_t)gncol[ni] * HW;
        if (res) v += res[o];
        Y[o] = v;
      }
    }
  }
}

// f32 weight -> bf16 K x N, optional transpose (src rows = N, cols = K)
__global__ void k_convw(const float* __restrict__ src, __bf16* __restrict__ dst,
                        int K, int N, int trans)
{
  int i = blockIdx.x * TB + threadIdx.x;
  if (i >= K * N) return;
  int k = i / N, n = i - k * N;
  float v = trans ? src[(size_t)n * K + k] : src[(size_t)k * N + n];
  dst[i] = f2bf(v);
}

// Direct 3x3 conv, pad=1, arbitrary stride
__global__ void k_conv3x3(const float* __restrict__ x, const float* __restrict__ w,
                          const float* __restrict__ bias, float* __restrict__ y,
                          int B, int Ci, int Hi, int Wi, int Co, int Ho, int Wo, int stride)
{
  int i = blockIdx.x * TB + threadIdx.x;
  int tot = B * Co * Ho * Wo;
  if (i >= tot) return;
  int wo = i % Wo; int t = i / Wo;
  int ho = t % Ho; t /= Ho;
  int co = t % Co; int b = t / Co;
  float s = bias ? bias[co] : 0.f;
  int hbase = ho * stride - 1, wbase = wo * stride - 1;
  for (int ci = 0; ci < Ci; ++ci) {
    const float* xp = x + ((size_t)b * Ci + ci) * Hi * Wi;
    const float* wp = w + ((size_t)co * Ci + ci) * 9;
    for (int kh = 0; kh < 3; ++kh) {
      int hh = hbase + kh; if (hh < 0 || hh >= Hi) continue;
      for (int kw = 0; kw < 3; ++kw) {
        int ww = wbase + kw; if (ww < 0 || ww >= Wi) continue;
        s += xp[hh * Wi + ww] * wp[kh * 3 + kw];
      }
    }
  }
  y[i] = s;
}

// Depthwise 3x3 (pad=1, stride=1, no bias) + SiLU
__global__ void k_dwconv_silu(const float* __restrict__ x, const float* __restrict__ w,
                              float* __restrict__ y, int B, int C, int H, int W)
{
  int i = blockIdx.x * TB + threadIdx.x;
  int tot = B * C * H * W;
  if (i >= tot) return;
  int ww = i % W; int t = i / W;
  int hh = t % H; t /= H;
  int c = t % C;  int b = t / C;
  const float* xp = x + ((size_t)b * C + c) * H * W;
  const float* wp = w + (size_t)c * 9;
  float s = 0.f;
  for (int kh = 0; kh < 3; ++kh) {
    int h2 = hh + kh - 1; if (h2 < 0 || h2 >= H) continue;
    for (int kw = 0; kw < 3; ++kw) {
      int w2 = ww + kw - 1; if (w2 < 0 || w2 >= W) continue;
      s += xp[h2 * W + w2] * wp[kh * 3 + kw];
    }
  }
  y[i] = siluf(s);
}

// Channel LayerNorm (per pixel), one wave32 per pixel; act=1 -> fused GELU
__global__ __launch_bounds__(TB)
void k_ln2d(const float* __restrict__ x, float* __restrict__ y,
            const float* __restrict__ g, const float* __restrict__ b,
            int Bsz, int C, int HW, int act)
{
  int wave = threadIdx.x >> 5, lane = threadIdx.x & 31;
  int pixel = blockIdx.x * 8 + wave;
  if (pixel >= Bsz * HW) return;
  int bb = pixel / HW, hw = pixel - bb * HW;
  const float* xp = x + (size_t)bb * C * HW + hw;
  float* yp = y + (size_t)bb * C * HW + hw;
  float s = 0.f, s2 = 0.f;
  for (int c = lane; c < C; c += 32) { float v = xp[(size_t)c * HW]; s += v; s2 += v * v; }
#pragma unroll
  for (int m = 16; m >= 1; m >>= 1) { s += __shfl_xor(s, m, 32); s2 += __shfl_xor(s2, m, 32); }
  float mean = s / C;
  float inv = rsqrtf(s2 / C - mean * mean + 1e-5f);
  for (int c = lane; c < C; c += 32) {
    float v = (xp[(size_t)c * HW] - mean) * inv * g[c] + b[c];
    if (act) v = geluf(v);
    yp[(size_t)c * HW] = v;
  }
}

// (B,C,H,W) -> (B,C,W,H) spatial transpose
__global__ void k_transp(const float* __restrict__ x, float* __restrict__ y,
                         int B, int C, int H, int W)
{
  int i = blockIdx.x * TB + threadIdx.x;
  int tot = B * C * H * W;
  if (i >= tot) return;
  int ww = i % W; int t = i / W;
  int hh = t % H; int bc = t / H;
  y[(size_t)bc * H * W + (size_t)ww * H + hh] = x[i];
}

// Selective scan, N_STATE=1: scalar recurrence per (b,c). rev=1 scans backwards
// and writes outputs at original positions == flip(ys) of the reference.
__global__ void k_scan(const float* __restrict__ x, const float* __restrict__ dt,
                       const float* __restrict__ Bv, const float* __restrict__ Cv,
                       const float* __restrict__ Alog, const float* __restrict__ Ds,
                       float* __restrict__ y,
                       int Bsz, int C, int L, int pbs, int rev)
{
  int i = blockIdx.x * TB + threadIdx.x;
  if (i >= Bsz * C) return;
  int c = i % C, b = i / C;
  float A  = -expf(Alog[c]);
  float Dv = Ds[c];
  const float* xp = x  + ((size_t)b * C + c) * L;
  const float* dp = dt + ((size_t)b * C + c) * L;
  const float* bp = Bv + (size_t)b * pbs;
  const float* cp = Cv + (size_t)b * pbs;
  float* yp = y + ((size_t)b * C + c) * L;
  float h = 0.f;
  for (int s = 0; s < L; ++s) {
    int l = rev ? (L - 1 - s) : s;
    float d = dp[l], xv = xp[l];
    h = expf(d * A) * h + d * bp[l] * xv;
    yp[l] = h * cp[l] + Dv * xv;
  }
}

// y = ys0 + ys2 + T(ys1 + ys3): lt = w*H + h maps the transposed branch back
__global__ void k_combine(const float* __restrict__ y0, const float* __restrict__ y1,
                          const float* __restrict__ y2, const float* __restrict__ y3,
                          float* __restrict__ out, int B, int C, int H, int W)
{
  int L = H * W;
  int i = blockIdx.x * TB + threadIdx.x;
  int tot = B * C * L;
  if (i >= tot) return;
  int l = i % L;
  size_t off = (size_t)(i / L) * L;
  int hh = l / W, ww = l - hh * W;
  int lt = ww * H + hh;
  out[off + l] = y0[off + l] + y2[off + l] + y1[off + lt] + y3[off + lt];
}

__global__ void k_meanpool(const float* __restrict__ feat, float* __restrict__ out,
                           int B, int O, int L)
{
  int i = blockIdx.x * TB + threadIdx.x;
  if (i >= B * O) return;
  const float* p = feat + (size_t)i * L;
  float s = 0.f;
  for (int l = 0; l < L; ++l) s += p[l];
  out[i] = s / L;
}

static inline int ceil_div(int a, int b) { return (a + b - 1) / b; }

extern "C" void kernel_launch(void* const* d_in, const int* in_sizes, int n_in,
                              void* d_out, int out_size, void* d_ws, size_t ws_size,
                              hipStream_t stream)
{
  (void)in_sizes; (void)n_in; (void)out_size; (void)ws_size;
  const int Bsz = 8;
  static const int DIMS[4]   = {96, 192, 384, 768};
  static const int DEPTHS[4] = {2, 2, 8, 2};

  // ---- flattened-parameter cursor (dict insertion order, depth first) ----
  auto P = [&](int i) -> const float* { return (const float*)d_in[i]; };
  int cur_idx = 1;                       // d_in[0] = x (8,3,224,224)
  int pe0 = cur_idx; cur_idx += 8;       // w1,b1,n1g,n1b,w2,b2,n2g,n2b
  int blk_base[4], down_base[4];
  for (int s = 0; s < 4; ++s) {
    blk_base[s] = cur_idx; cur_idx += DEPTHS[s] * 18;  // 18 leaves per block
    if (s < 3) { down_base[s] = cur_idx; cur_idx += 4; } else down_base[s] = -1;
  }
  int outn = cur_idx; cur_idx += 2;
  int headw = cur_idx, headb = cur_idx + 1;

  // ---- bump allocator over d_ws (deterministic each call) ----
  size_t off = 0;
  auto allocF = [&](size_t n) -> float* {
    float* p = (float*)((char*)d_ws + off);
    off += ((n * sizeof(float) + 255) & ~(size_t)255);
    return p;
  };
  const size_t ACT = (size_t)Bsz * 96 * 3136;   // max B*C*L across stages
  float* cur  = allocF(ACT);
  float* nxt  = allocF(ACT);
  float* lnb  = allocF(ACT);
  float* hbuf = allocF(ACT);
  float* hcb  = allocF(ACT);
  float* xtb  = allocF(ACT);
  float* dts  = allocF(ACT);
  float* ysb[4]; for (int k = 0; k < 4; ++k) ysb[k] = allocF(ACT);
  float* ycmb = allocF(ACT);
  float* mlpb = allocF(ACT * 4);
  float* xdbl = allocF((size_t)Bsz * 16 * 3136);  // B*(R+2)*L max (stage0: 8*8*3136)
  float* pebuf = allocF((size_t)Bsz * 48 * 112 * 112);
  __bf16* wbf = (__bf16*)((char*)d_ws + off);     // weight staging (max 768*3072)
  off += (((size_t)768 * 3072 * sizeof(__bf16)) + 255) & ~(size_t)255;

  // ---- launch helpers ----
  auto convw = [&](const float* src, int K, int N, int trans) {
    int tot = K * N;
    k_convw<<<ceil_div(tot, TB), TB, 0, stream>>>(src, wbf, K, N, trans);
  };
  auto gemm = [&](const float* X, const float* bias, const float* res, float* Y,
                  int M, int N, int K, int HW, int xbs, int ybs, int act) {
    dim3 g(ceil_div(M, BM), ceil_div(N, BN));
    k_gemm_wmma<<<g, TB, 0, stream>>>(X, wbf, bias, res, Y, M, N, K, HW, xbs, ybs, act);
  };
  auto ln = [&](const float* x, float* y, const float* g, const float* b,
                int C, int HW, int act) {
    k_ln2d<<<ceil_div(Bsz * HW, 8), TB, 0, stream>>>(x, y, g, b, Bsz, C, HW, act);
  };

  // ---- patch embed ----
  {
    const float* x = P(0);
    int tot = Bsz * 48 * 112 * 112;
    k_conv3x3<<<ceil_div(tot, TB), TB, 0, stream>>>(x, P(pe0 + 0), P(pe0 + 1), pebuf,
                                                    Bsz, 3, 224, 224, 48, 112, 112, 2);
    ln(pebuf, pebuf, P(pe0 + 2), P(pe0 + 3), 48, 112 * 112, 1);   // gelu(ln)
    tot = Bsz * 96 * 56 * 56;
    k_conv3x3<<<ceil_div(tot, TB), TB, 0, stream>>>(pebuf, P(pe0 + 4), P(pe0 + 5), cur,
                                                    Bsz, 48, 112, 112, 96, 56, 56, 2);
    ln(cur, cur, P(pe0 + 6), P(pe0 + 7), 96, 56 * 56, 0);
  }

  // ---- stages ----
  int H = 56, W = 56;
  for (int s = 0; s < 4; ++s) {
    int C = DIMS[s], L = H * W, R = (C + 15) / 16, E = R + 2;
    for (int d = 0; d < DEPTHS[s]; ++d) {
      int pb = blk_base[s] + d * 18;
      // --- ss2d branch ---
      ln(cur, lnb, P(pb + 0), P(pb + 1), C, L, 0);
      convw(P(pb + 2), C, C, 0);                                   // in_w (C,C)
      gemm(lnb, nullptr, nullptr, hbuf, Bsz * L, C, C, L, C * L, C * L, 0);
      {
        int tot = Bsz * C * L;
        k_dwconv_silu<<<ceil_div(tot, TB), TB, 0, stream>>>(hbuf, P(pb + 3), hcb, Bsz, C, H, W);
        k_transp<<<ceil_div(tot, TB), TB, 0, stream>>>(hcb, xtb, Bsz, C, H, W);
      }
      for (int k = 0; k < 4; ++k) {
        const float* src = (k & 1) ? xtb : hcb;                    // k=0,2: xf; k=1,3: xt
        convw(P(pb + 4) + (size_t)k * E * C, C, E, 1);             // xproj[k] (E,C)^T
        gemm(src, nullptr, nullptr, xdbl, Bsz * L, E, C, L, C * L, E * L, 0);
        convw(P(pb + 5) + (size_t)k * C * R, R, C, 1);             // dt_w[k] (C,R)^T
        gemm(xdbl, P(pb + 6) + (size_t)k * C, nullptr, dts,
             Bsz * L, C, R, L, E * L, C * L, 2 /*softplus*/);
        k_scan<<<ceil_div(Bsz * C, TB), TB, 0, stream>>>(
            src, dts, xdbl + (size_t)R * L, xdbl + (size_t)(R + 1) * L,
            P(pb + 7) + (size_t)k * C, P(pb + 8) + (size_t)k * C, ysb[k],
            Bsz, C, L, E * L, (k >= 2) ? 1 : 0);
      }
      {
        int tot = Bsz * C * L;
        k_combine<<<ceil_div(tot, TB), TB, 0, stream>>>(ysb[0], ysb[1], ysb[2], ysb[3],
                                                        ycmb, Bsz, C, H, W);
      }
      ln(ycmb, ycmb, P(pb + 9), P(pb + 10), C, L, 0);              // onorm
      convw(P(pb + 11), C, C, 0);                                  // out_w
      gemm(ycmb, nullptr, cur /*residual*/, nxt, Bsz * L, C, C, L, C * L, C * L, 0);
      std::swap(cur, nxt);
      // --- MLP branch ---
      ln(cur, lnb, P(pb + 12), P(pb + 13), C, L, 0);
      convw(P(pb + 14), C, 4 * C, 0);                              // fc1
      gemm(lnb, P(pb + 15), nullptr, mlpb, Bsz * L, 4 * C, C, L, C * L, 4 * C * L, 1 /*gelu*/);
      convw(P(pb + 16), 4 * C, C, 0);                              // fc2
      gemm(mlpb, P(pb + 17), cur /*residual*/, nxt, Bsz * L, C, 4 * C, L, 4 * C * L, C * L, 0);
      std::swap(cur, nxt);
    }
    if (s < 3) {
      int db = down_base[s];
      int Ho = H / 2, Wo = W / 2, Co = 2 * C;
      int tot = Bsz * Co * Ho * Wo;
      k_conv3x3<<<ceil_div(tot, TB), TB, 0, stream>>>(cur, P(db + 0), P(db + 1), nxt,
                                                      Bsz, C, H, W, Co, Ho, Wo, 2);
      std::swap(cur, nxt);
      ln(cur, cur, P(db + 2), P(db + 3), Co, Ho * Wo, 0);
      H = Ho; W = Wo;
    }
  }

  // ---- head ----
  {
    int C = 768, L = H * W;                                        // 7*7 = 49
    ln(cur, cur, P(outn), P(outn + 1), C, L, 0);
    convw(P(headw), C, 2048, 1);                                   // head_w (2048,768)^T
    float* feat = (float*)d_out;
    gemm(cur, P(headb), nullptr, feat, Bsz * L, 2048, C, L, C * L, 2048 * L, 0);
    float* pooled = feat + (size_t)Bsz * 2048 * L;
    k_meanpool<<<ceil_div(Bsz * 2048, TB), TB, 0, stream>>>(feat, pooled, Bsz, 2048, L);
  }
}
